// WePred_83330955477484
// MI455X (gfx1250) — matrix operand order repaired and checked by reading, not compiled, not run on hardware
//
#include <hip/hip_runtime.h>
#include <hip/hip_bf16.h>
#include <math.h>

typedef __attribute__((ext_vector_type(16))) _Float16 v16h;
typedef __attribute__((ext_vector_type(8)))  _Float16 v8h;
typedef __attribute__((ext_vector_type(8)))  float    v8f;

#define HEADS  4
#define DMODEL 128
#define TPB    256

// ---------------------------------------------------------------- utilities
__global__ void k_fill_f32(float* __restrict__ p, float v, int n) {
    int i = blockIdx.x * blockDim.x + threadIdx.x;
    if (i < n) p[i] = v;
}

__global__ void k_gather_f16(const float* __restrict__ tab, const int* __restrict__ idx,
                             _Float16* __restrict__ out, int rows, int cols) {
    int i = blockIdx.x * blockDim.x + threadIdx.x;
    if (i >= rows * cols) return;
    int r = i / cols, c = i - r * cols;
    out[i] = (_Float16)tab[(size_t)idx[r] * cols + c];
}

// weight convert + transpose: in[K,128] f32 -> out[128,K] f16 (column-major B)
// makes every WMMA B-fragment a contiguous 32-byte run -> 2x global_load_b128
__global__ void k_cvt_f16_T(const float* __restrict__ in, _Float16* __restrict__ out, int K) {
    int i = blockIdx.x * blockDim.x + threadIdx.x;          // K*128
    if (i >= K * DMODEL) return;
    int k = i / DMODEL, c = i - k * DMODEL;
    out[(size_t)c * K + k] = (_Float16)in[i];
}

// ------------------------------------------------- WMMA GEMM: C[M,128] = A[M,K] @ Bt
// A row-major [M,K] f16; Bt column-major [128,K] f16 (pre-transposed weights).
// 256 threads = 8 waves; wave w owns output columns [16w,16w+16).
// B fragments (M-invariant) preloaded into registers, reused across MT row sub-tiles.
template <int K, int MT>
__global__ void __launch_bounds__(256)
k_gemm_wmma(const _Float16* __restrict__ A, const _Float16* __restrict__ Bt,
            float* __restrict__ C, int M) {
    const int N = DMODEL;
    constexpr int KC = K / 32;             // K-chunks of 32
    int lane  = threadIdx.x & 31;
    int wave  = threadIdx.x >> 5;          // 0..7 -> n tile
    int l15   = lane & 15;
    int hi    = lane >> 4;                 // 0 or 1
    int nBase = wave * 16;
    int koffA = hi * 16;                   // A: lanes16-31 hold K 16-31
    int koffB = hi * 16;                   // B: lanes16-31 hold K 16-31

    union Frag { v16h v; v8h h[2]; };

    // ---- preload all B fragments: lane's 16 halves are K-contiguous in Bt
    Frag bf[KC];
    const _Float16* Bcol = Bt + (size_t)(nBase + l15) * K + koffB;
#pragma unroll
    for (int kc = 0; kc < KC; ++kc) {
        bf[kc].h[0] = *(const v8h*)(Bcol + kc * 32);
        bf[kc].h[1] = *(const v8h*)(Bcol + kc * 32 + 8);
    }

#pragma unroll
    for (int t = 0; t < MT; ++t) {
        int mBase = (blockIdx.x * MT + t) * 16;
        if (mBase >= M) break;                                  // uniform exit
        int mRow = mBase + l15; if (mRow >= M) mRow = M - 1;    // clamp reads
        const _Float16* Arow = A + (size_t)mRow * K + koffA;

        v8f acc = {};
#pragma unroll
        for (int kc = 0; kc < KC; ++kc) {
            Frag a;                        // two contiguous 16B runs per lane
            a.h[0] = *(const v8h*)(Arow + kc * 32);
            a.h[1] = *(const v8h*)(Arow + kc * 32 + 8);
            acc = __builtin_amdgcn_wmma_f32_16x16x32_f16(
                /*neg_a=*/false, a.v, /*neg_b=*/false, bf[kc].v,
                /*c_mod=*/(short)0, acc, /*reuse_a=*/false, /*reuse_b=*/false);
        }

        // C/D 16x16 f32: VGPR r -> row r (lanes 0-15) / row r+8 (lanes 16-31)
        int mOut = mBase + hi * 8;
        if (mBase + 16 <= M) {             // uniform full-tile fast path
#pragma unroll
            for (int r = 0; r < 8; ++r)
                C[(size_t)(mOut + r) * N + nBase + l15] = acc[r];
        } else {
#pragma unroll
            for (int r = 0; r < 8; ++r)
                if (mOut + r < M)
                    C[(size_t)(mOut + r) * N + nBase + l15] = acc[r];
        }
    }
}

// ---------------------------------------- per-head column sums of Wq, Wk, We -> wsum[3][128][4]
__global__ void k_colsum3(const float* __restrict__ Wq, const float* __restrict__ Wk,
                          const float* __restrict__ We, float* __restrict__ wsum) {
    int i = blockIdx.x * blockDim.x + threadIdx.x;          // 3*128*4 = 1536
    if (i >= 3 * DMODEL * HEADS) return;
    int which = i / (DMODEL * HEADS);
    int rem   = i - which * (DMODEL * HEADS);
    int d = rem / HEADS, h = rem - d * HEADS;
    const float* W = which == 0 ? Wq : (which == 1 ? Wk : We);
    float s = 0.f;
#pragma unroll
    for (int c = 0; c < 32; ++c) s += W[d * DMODEL + h * 32 + c];
    wsum[i] = s;
}

// qs[n,h] = x[n,:] @ wsumQ[:,h] ; ks likewise (skinny GEMV, ~0.1 GFLOP total)
__global__ void k_qsks(const float* __restrict__ x, const float* __restrict__ wsum,
                       float* __restrict__ qs, float* __restrict__ ks, int Nn) {
    int i = blockIdx.x * blockDim.x + threadIdx.x;          // Nn*8
    if (i >= Nn * 8) return;
    int n = i >> 3, j = i & 7;
    int which = j >> 2, h = j & 3;
    const float* ws = wsum + which * (DMODEL * HEADS);
    const float* xr = x + (size_t)n * DMODEL;
    float s = 0.f;
    for (int d = 0; d < DMODEL; ++d) s += xr[d] * ws[d * HEADS + h];
    if (which == 0) qs[n * HEADS + h] = s; else ks[n * HEADS + h] = s;
}

// es[r,h] = rel_emb[r,:] @ colsum_h(We)
__global__ void k_es(const float* __restrict__ rel, const float* __restrict__ wsumE,
                     float* __restrict__ es, int R) {
    int i = blockIdx.x * blockDim.x + threadIdx.x;          // R*4
    if (i >= R * HEADS) return;
    int r = i >> 2, h = i & 3;
    const float* rr = rel + (size_t)r * DMODEL;
    float s = 0.f;
    for (int d = 0; d < DMODEL; ++d) s += rr[d] * wsumE[d * HEADS + h];
    es[i] = s;
}

// ---------------------------------------------------------------- edge passes
__device__ inline void atomicMaxF(float* addr, float val) {
    if (val >= 0.f) atomicMax((int*)addr, __float_as_int(val));
    else            atomicMin((unsigned int*)addr, __float_as_uint(val));
}

__global__ void k_edge_alpha(const int* __restrict__ src, const int* __restrict__ dst,
                             const int* __restrict__ etype, const float* __restrict__ ew,
                             const float* __restrict__ qs, const float* __restrict__ ks,
                             const float* __restrict__ es, const float* __restrict__ gate,
                             float* __restrict__ alpha, float* __restrict__ mbuf, int E) {
    int e = blockIdx.x * blockDim.x + threadIdx.x;
    if (e >= E) return;
    int s = src[e], d = dst[e], t = etype[e];
    float w = ew[e];
    const float rs = 0.17677669529663687f;   // 1/sqrt(32)
#pragma unroll
    for (int h = 0; h < HEADS; ++h) {
        float a = (qs[d * HEADS + h] + ks[s * HEADS + h] + es[t * HEADS + h]) * rs
                + gate[h] * w;
        a = a > 0.f ? a : 0.2f * a;          // leaky_relu(0.2)
        alpha[(size_t)e * HEADS + h] = a;
        atomicMaxF(&mbuf[d * HEADS + h], a);
    }
}

__global__ void k_edge_exp(const int* __restrict__ dst, float* __restrict__ alpha,
                           const float* __restrict__ mbuf, float* __restrict__ sbuf, int E) {
    int e = blockIdx.x * blockDim.x + threadIdx.x;
    if (e >= E) return;
    int d = dst[e];
#pragma unroll
    for (int h = 0; h < HEADS; ++h) {
        float ex = __expf(alpha[(size_t)e * HEADS + h] - mbuf[d * HEADS + h]);
        alpha[(size_t)e * HEADS + h] = ex;
        atomicAdd(&sbuf[d * HEADS + h], ex);
    }
}

// one thread per (edge, 4-column group): coalesced float4 gather of v[src], f32 atomic scatter
__global__ void k_edge_aggr(const int* __restrict__ src, const int* __restrict__ dst,
                            const float* __restrict__ v, const float* __restrict__ alpha,
                            const float* __restrict__ sbuf, float* __restrict__ acc, int E) {
    int tid = blockIdx.x * blockDim.x + threadIdx.x;        // E*32
    if (tid >= E * 32) return;
    int e  = tid >> 5;
    int c4 = (tid & 31) << 2;
    int h  = c4 >> 5;
    int s = src[e], d = dst[e];
    float w = alpha[(size_t)e * HEADS + h] / (sbuf[d * HEADS + h] + 1e-16f);
    const float4 vv = *(const float4*)(v + (size_t)s * DMODEL + c4);
    float* out = acc + (size_t)d * DMODEL + c4;
    atomicAdd(out + 0, vv.x * w);
    atomicAdd(out + 1, vv.y * w);
    atomicAdd(out + 2, vv.z * w);
    atomicAdd(out + 3, vv.w * w);
}

// out = in + bias [; elu] ; optionally mirror to f16 for the next WMMA GEMM
__global__ void k_bias_act(const float* __restrict__ in, const float* __restrict__ bias,
                           float* __restrict__ out_f32, _Float16* __restrict__ out_f16,
                           int n, int do_elu, int do_f16) {
    int i = blockIdx.x * blockDim.x + threadIdx.x;
    if (i >= n) return;
    int c = i & (DMODEL - 1);
    float v = in[i] + bias[c];
    if (do_elu) v = v > 0.f ? v : expm1f(v);
    out_f32[i] = v;
    if (do_f16) out_f16[i] = (_Float16)v;
}

// ---------------------------------------------------------------- launcher
extern "C" void kernel_launch(void* const* d_in, const int* in_sizes, int n_in,
                              void* d_out, int out_size, void* d_ws, size_t ws_size,
                              hipStream_t stream) {
    const int*   entity    = (const int*)  d_in[0];
    const int*   eidx      = (const int*)  d_in[1];   // [2,E]
    const int*   etype     = (const int*)  d_in[2];
    const float* ew        = (const float*)d_in[3];
    const float* ent_table = (const float*)d_in[4];
    const float* proj_w    = (const float*)d_in[5];
    const float* proj_b    = (const float*)d_in[6];
    const float* rel_emb   = (const float*)d_in[7];
    const float* Wq        = (const float*)d_in[8];
    const float* Wk        = (const float*)d_in[9];
    const float* Wv        = (const float*)d_in[10];
    const float* We        = (const float*)d_in[11];
    const float* gate      = (const float*)d_in[12];  // [L,H]
    const float* cbias     = (const float*)d_in[13];  // [L,D]

    const int N = in_sizes[0];
    const int E = in_sizes[3];
    const int D = DMODEL;
    const int F = in_sizes[5] / D;   // 256
    const int R = in_sizes[7] / D;   // 200
    const int L = 2;
    (void)n_in; (void)out_size; (void)ws_size;

    // workspace bump allocator (~125 MB total)
    char* wsb = (char*)d_ws;
    size_t off = 0;
    auto alloc = [&](size_t bytes) -> char* {
        char* p = wsb + off;
        off = (off + bytes + 255) & ~(size_t)255;
        return p;
    };
    _Float16* ent_h = (_Float16*)alloc((size_t)N * F * 2);
    _Float16* w_h   = (_Float16*)alloc((size_t)F * D * 2);  // transposed weights [128][K]
    float*    x32   = (float*)   alloc((size_t)N * D * 4);
    _Float16* x16   = (_Float16*)alloc((size_t)N * D * 2);
    float*    vbuf  = (float*)   alloc((size_t)N * D * 4);
    float*    accb  = (float*)   alloc((size_t)N * D * 4);
    float*    alpha = (float*)   alloc((size_t)E * HEADS * 4);
    float*    mbuf  = (float*)   alloc((size_t)N * HEADS * 4);
    float*    sbuf  = (float*)   alloc((size_t)N * HEADS * 4);
    float*    qs    = (float*)   alloc((size_t)N * HEADS * 4);
    float*    ks    = (float*)   alloc((size_t)N * HEADS * 4);
    float*    wsum  = (float*)   alloc((size_t)3 * D * HEADS * 4);
    float*    esb   = (float*)   alloc((size_t)R * HEADS * 4);

    const int* srcp = eidx;
    const int* dstp = eidx + E;

    auto cdiv = [](long a, long b) { return (int)((a + b - 1) / b); };
    constexpr int MT = 5;            // 50000 = 625 * (16*5): no tail in practice

    // x = ent_table[entity] @ proj_w + proj_b
    k_gather_f16<<<cdiv((long)N * F, TPB), TPB, 0, stream>>>(ent_table, entity, ent_h, N, F);
    k_cvt_f16_T <<<cdiv((long)F * D, TPB), TPB, 0, stream>>>(proj_w, w_h, F);
    k_gemm_wmma<256, MT><<<cdiv(N, 16 * MT), TPB, 0, stream>>>(ent_h, w_h, accb, N);
    k_bias_act  <<<cdiv((long)N * D, TPB), TPB, 0, stream>>>(accb, proj_b, x32, x16, N * D, 0, 1);

    for (int l = 0; l < L; ++l) {
        const float* Wql = Wq + (size_t)l * D * D;
        const float* Wkl = Wk + (size_t)l * D * D;
        const float* Wvl = Wv + (size_t)l * D * D;
        const float* Wel = We + (size_t)l * D * D;
        const float* gl  = gate  + (size_t)l * HEADS;
        const float* bl  = cbias + (size_t)l * D;

        // per-head attention scalars (algebraic reduction of q/k/e GEMMs)
        k_colsum3<<<cdiv(3 * D * HEADS, TPB), TPB, 0, stream>>>(Wql, Wkl, Wel, wsum);
        k_qsks   <<<cdiv((long)N * 8,   TPB), TPB, 0, stream>>>(x32, wsum, qs, ks, N);
        k_es     <<<cdiv(R * HEADS,     TPB), TPB, 0, stream>>>(rel_emb, wsum + 2 * D * HEADS, esb, R);

        // v = x @ Wv  (WMMA)
        k_cvt_f16_T<<<cdiv(D * D, TPB), TPB, 0, stream>>>(Wvl, w_h, D);
        k_gemm_wmma<128, MT><<<cdiv(N, 16 * MT), TPB, 0, stream>>>(x16, w_h, vbuf, N);

        // segment softmax + aggregation
        k_fill_f32<<<cdiv((long)N * HEADS, TPB), TPB, 0, stream>>>(mbuf, -INFINITY, N * HEADS);
        k_fill_f32<<<cdiv((long)N * HEADS, TPB), TPB, 0, stream>>>(sbuf, 0.f,       N * HEADS);
        k_fill_f32<<<cdiv((long)N * D,     TPB), TPB, 0, stream>>>(accb, 0.f,       N * D);

        k_edge_alpha<<<cdiv(E, TPB),            TPB, 0, stream>>>(srcp, dstp, etype, ew, qs, ks, esb, gl, alpha, mbuf, E);
        k_edge_exp  <<<cdiv(E, TPB),            TPB, 0, stream>>>(dstp, alpha, mbuf, sbuf, E);
        k_edge_aggr <<<cdiv((long)E * 32, TPB), TPB, 0, stream>>>(srcp, dstp, vbuf, alpha, sbuf, accb, E);

        if (l != L - 1)
            k_bias_act<<<cdiv((long)N * D, TPB), TPB, 0, stream>>>(accb, bl, x32, x16, N * D, 1, 1);
        else
            k_bias_act<<<cdiv((long)N * D, TPB), TPB, 0, stream>>>(accb, bl, (float*)d_out, x16, N * D, 0, 0);
    }
}
